// gcn_tuple_Net_67508295958859
// MI455X (gfx1250) — matrix operand order repaired and checked by reading, not compiled, not run on hardware
//
#include <hip/hip_runtime.h>

// Problem constants (match reference)
#define NN 100000
#define EE 1600000
#define GG 512

typedef __attribute__((ext_vector_type(16))) _Float16 v16h;
typedef __attribute__((ext_vector_type(8)))  float    v8f;

static constexpr int LDSS = 136; // f16 element stride: 16B-aligned fragments, 68-dword rows

// ---------------- elementwise / setup kernels ----------------

__global__ __launch_bounds__(256) void k_fill1(float* p, int n) {
  int i = blockIdx.x * 256 + threadIdx.x;
  if (i < n) p[i] = 1.0f;
}

__global__ __launch_bounds__(256) void k_zero(float* p, int n) {
  int i = blockIdx.x * 256 + threadIdx.x;
  if (i < n) p[i] = 0.0f;
}

__global__ __launch_bounds__(256) void k_deg_acc(const int* __restrict__ ei,
                                                 float* __restrict__ deg) {
  int e = blockIdx.x * 256 + threadIdx.x;
  if (e < EE) atomicAdd(&deg[ei[EE + e]], 1.0f);   // dst side
}

__global__ __launch_bounds__(256) void k_rsqrt(float* p, int n) {
  int i = blockIdx.x * 256 + threadIdx.x;
  if (i < n) p[i] = rsqrtf(p[i]);                  // deg >= 1 always (self loop)
}

// ---------------- WMMA GEMM: out[M,64] = A[M,K] @ W[K,64] (+bias)(+relu) ----
// block: 256 thr = 8 waves; block tile 32x64; wave tile 16x16; K in {64,128}

__global__ __launch_bounds__(256) void k_gemm(const float* __restrict__ A,
                                              const float* __restrict__ W,
                                              const float* __restrict__ bias,
                                              float* __restrict__ out,
                                              int K, int kbits, int relu)
{
  __shared__ __align__(16) _Float16 As[32 * LDSS];
  __shared__ __align__(16) _Float16 Bs[64 * LDSS];
  const int tid = threadIdx.x;
  const long rowBase = (long)blockIdx.x * 32;

  // stage A tile (fp32 -> f16), coalesced over k
  for (int i = tid; i < (32 << kbits); i += 256) {
    int r = i >> kbits, k = i & (K - 1);
    As[r * LDSS + k] = (_Float16)A[(rowBase + r) * K + k];
  }
  // stage W transposed: Bs[n][k] = W[k][n]
  for (int i = tid; i < (K << 6); i += 256) {
    int k = i >> 6, n = i & 63;
    Bs[n * LDSS + k] = (_Float16)W[i];
  }
  __syncthreads();

  const int wave = tid >> 5, lane = tid & 31;
  const int rowTile = (wave >> 2) << 4;       // 0 / 16
  const int colTile = (wave & 3) << 4;        // 0,16,32,48
  const int l15 = lane & 15, hi = lane >> 4;  // hi: K-half select
  const int m = rowTile + l15;                // A row per lane
  const int n = colTile + l15;                // B/C col per lane

  v8f c = {};
  for (int k0 = 0; k0 < K; k0 += 32) {
    v16h a, b;
    // A frag: elems 0-7 = K[k0+hi*8 .. +7], elems 8-15 = K[k0+16+hi*8 .. +7]
    const _Float16* ap = &As[m * LDSS + k0 + hi * 8];
    ((uint4*)&a)[0] = *(const uint4*)(ap);
    ((uint4*)&a)[1] = *(const uint4*)(ap + 16);
    // B frag: elems 0-7 = K[k0+hi*8 .. +7], elems 8-15 = K[k0+16+hi*8 .. +7] for column n
    const _Float16* bp = &Bs[n * LDSS + k0 + hi * 8];
    ((uint4*)&b)[0] = *(const uint4*)(bp);
    ((uint4*)&b)[1] = *(const uint4*)(bp + 16);
    c = __builtin_amdgcn_wmma_f32_16x16x32_f16(false, a, false, b,
                                               (short)0, c, false, false);
  }

  const float bv = bias ? bias[n] : 0.0f;
  float* op = out + rowBase * 64 + n;
#pragma unroll
  for (int r = 0; r < 8; ++r) {
    int mm = rowTile + hi * 8 + r;            // C layout: VGPR r -> M=r / r+8
    float v = c[r] + bv;
    if (relu) v = fmaxf(v, 0.0f);
    op[(long)mm * 64] = v;
  }
}

// ---------------- GCN propagation ----------------

// self-loop contribution: out[n,:] = dis[n]^2 * t[n,:]
__global__ __launch_bounds__(256) void k_self(const float* __restrict__ dis,
                                              const float* __restrict__ t,
                                              float* __restrict__ outp) {
  long i = (long)blockIdx.x * 256 + threadIdx.x;
  if (i < (long)NN * 64) {
    int node = (int)(i >> 6);
    float w = dis[node];
    outp[i] = w * w * t[i];
  }
}

// one wave per edge; each lane covers 2 channels
__global__ __launch_bounds__(256) void k_scatter(const int* __restrict__ ei,
                                                 const float* __restrict__ dis,
                                                 const float* __restrict__ t,
                                                 float* __restrict__ outp) {
  int e = blockIdx.x * 8 + (threadIdx.x >> 5);
  if (e >= EE) return;
  int lane = threadIdx.x & 31;
  int s = ei[e];
  int d = ei[EE + e];
  float w = dis[s] * dis[d];
  const float* ts = t + (long)s * 64;
  float* od = outp + (long)d * 64;
  atomicAdd(&od[lane],      w * ts[lane]);
  atomicAdd(&od[lane + 32], w * ts[lane + 32]);
}

// H[n, coff + c] = relu(agg[n,c] + bias[c])   (concat of the two branches)
__global__ __launch_bounds__(256) void k_bias_relu_cat(const float* __restrict__ agg,
                                                       const float* __restrict__ bias,
                                                       float* __restrict__ H, int coff) {
  long i = (long)blockIdx.x * 256 + threadIdx.x;
  if (i < (long)NN * 64) {
    int node = (int)(i >> 6), c = (int)(i & 63);
    H[(long)node * 128 + coff + c] = fmaxf(agg[i] + bias[c], 0.0f);
  }
}

// ---------------- pooling + final linear ----------------

__global__ __launch_bounds__(256) void k_pool(const float* __restrict__ g,
                                              const int* __restrict__ batch,
                                              float* __restrict__ pooled) {
  long i = (long)blockIdx.x * 256 + threadIdx.x;
  if (i < (long)NN * 64) {
    int node = (int)(i >> 6), c = (int)(i & 63);
    atomicAdd(&pooled[(long)batch[node] * 64 + c], g[i]);
  }
}

__global__ __launch_bounds__(256) void k_final(const float* __restrict__ pooled,
                                               const float* __restrict__ lw,
                                               const float* __restrict__ lb,
                                               float* __restrict__ out) {
  int g = blockIdx.x * 256 + threadIdx.x;
  if (g < GG) {
    float acc = 0.0f;
#pragma unroll
    for (int c = 0; c < 64; ++c) acc += pooled[g * 64 + c] * lw[c];
    out[g] = acc + lb[0];
  }
}

// ---------------- launcher ----------------

extern "C" void kernel_launch(void* const* d_in, const int* in_sizes, int n_in,
                              void* d_out, int out_size, void* d_ws, size_t ws_size,
                              hipStream_t stream) {
  const float* x    = (const float*)d_in[0];
  const int*   bat  = (const int*)  d_in[1];
  const int*   ei0  = (const int*)  d_in[2];
  const int*   ei1  = (const int*)  d_in[3];
  const float* W1_0 = (const float*)d_in[4];
  const float* b1_0 = (const float*)d_in[5];
  const float* W1_1 = (const float*)d_in[6];
  const float* b1_1 = (const float*)d_in[7];
  const float* W2_0 = (const float*)d_in[8];
  const float* b2_0 = (const float*)d_in[9];
  const float* W2_1 = (const float*)d_in[10];
  const float* b2_1 = (const float*)d_in[11];
  const float* m1w1 = (const float*)d_in[12];
  const float* m1b1 = (const float*)d_in[13];
  const float* m1w2 = (const float*)d_in[14];
  const float* m1b2 = (const float*)d_in[15];
  const float* m2w1 = (const float*)d_in[16];
  const float* m2b1 = (const float*)d_in[17];
  const float* m2w2 = (const float*)d_in[18];
  const float* m2b2 = (const float*)d_in[19];
  const float* linw = (const float*)d_in[20];
  const float* linb = (const float*)d_in[21];
  float* out = (float*)d_out;

  // workspace layout (fp32): 450*N + 32768 floats ~= 180 MB
  float* ws     = (float*)d_ws;
  float* dis0   = ws;
  float* dis1   = dis0 + NN;
  float* P      = dis1 + NN;                 // N*64
  float* Q      = P + (size_t)NN * 64;       // N*64
  float* R      = Q + (size_t)NN * 64;       // N*64
  float* S      = R + (size_t)NN * 64;       // N*64
  float* H      = S + (size_t)NN * 64;       // N*128
  float* Mb     = H + (size_t)NN * 128;      // N*64
  float* pooled = Mb + (size_t)NN * 64;      // G*64

  const int B256    = 256;
  const int gN2     = (2 * NN + 255) / 256;      // both dis arrays (contiguous)
  const int gE      = (EE + 255) / 256;
  const int gEW     = (EE + 7) / 8;              // wave per edge
  const int gNV     = (int)(((long)NN * 64 + 255) / 256);
  const int gGemm   = NN / 32;                   // 3125, exact

  // degrees -> dis = rsqrt(deg)
  k_fill1  <<<gN2, B256, 0, stream>>>(dis0, 2 * NN);
  k_deg_acc<<<gE,  B256, 0, stream>>>(ei0, dis0);
  k_deg_acc<<<gE,  B256, 0, stream>>>(ei1, dis1);
  k_rsqrt  <<<gN2, B256, 0, stream>>>(dis0, 2 * NN);

  // layer 1: t = x @ W1_b
  k_gemm<<<gGemm, B256, 0, stream>>>(x, W1_0, nullptr, P, 128, 7, 0);
  k_gemm<<<gGemm, B256, 0, stream>>>(x, W1_1, nullptr, Q, 128, 7, 0);
  k_self   <<<gNV, B256, 0, stream>>>(dis0, P, R);
  k_scatter<<<gEW, B256, 0, stream>>>(ei0, dis0, P, R);
  k_self   <<<gNV, B256, 0, stream>>>(dis1, Q, S);
  k_scatter<<<gEW, B256, 0, stream>>>(ei1, dis1, Q, S);
  k_bias_relu_cat<<<gNV, B256, 0, stream>>>(R, b1_0, H, 0);
  k_bias_relu_cat<<<gNV, B256, 0, stream>>>(S, b1_1, H, 64);

  // mlp_1
  k_gemm<<<gGemm, B256, 0, stream>>>(H, m1w1, m1b1, P, 128, 7, 1);
  k_gemm<<<gGemm, B256, 0, stream>>>(P, m1w2, m1b2, Mb, 64, 6, 0);

  // layer 2
  k_gemm<<<gGemm, B256, 0, stream>>>(Mb, W2_0, nullptr, P, 64, 6, 0);
  k_gemm<<<gGemm, B256, 0, stream>>>(Mb, W2_1, nullptr, Q, 64, 6, 0);
  k_self   <<<gNV, B256, 0, stream>>>(dis0, P, R);
  k_scatter<<<gEW, B256, 0, stream>>>(ei0, dis0, P, R);
  k_self   <<<gNV, B256, 0, stream>>>(dis1, Q, S);
  k_scatter<<<gEW, B256, 0, stream>>>(ei1, dis1, Q, S);
  k_bias_relu_cat<<<gNV, B256, 0, stream>>>(R, b2_0, H, 0);
  k_bias_relu_cat<<<gNV, B256, 0, stream>>>(S, b2_1, H, 64);

  // mlp_2
  k_gemm<<<gGemm, B256, 0, stream>>>(H, m2w1, m2b1, P, 128, 7, 1);
  k_gemm<<<gGemm, B256, 0, stream>>>(P, m2w2, m2b2, Mb, 64, 6, 0);

  // global add pool + final linear
  k_zero <<<(GG * 64 + 255) / 256, B256, 0, stream>>>(pooled, GG * 64);
  k_pool <<<gNV, B256, 0, stream>>>(Mb, bat, pooled);
  k_final<<<(GG + 255) / 256, B256, 0, stream>>>(pooled, linw, linb, out);
}
